// FlashInferCausalGroupedQueryPageAttentionHandler_57870389346728
// MI455X (gfx1250) — matrix operand order repaired and checked by reading, not compiled, not run on hardware
//
#include <hip/hip_runtime.h>
#include <stdint.h>

#define DEVFN __device__ __forceinline__

typedef __attribute__((ext_vector_type(16))) __bf16 v16bf;
typedef __attribute__((ext_vector_type(8)))  __bf16 v8bf;
typedef __attribute__((ext_vector_type(8)))  float  v8f;
typedef __attribute__((ext_vector_type(4)))  float  f32x4;
typedef __attribute__((ext_vector_type(4)))  int    v4i;

namespace {
constexpr int kB    = 32;
constexpr int kQ    = 16;
constexpr int kHq   = 32;
constexpr int kHkv  = 8;
constexpr int kD    = 128;
constexpr int kPage = 16;
constexpr int kPPS  = 256;
constexpr int kKtot = kPage * kPPS;       // 4096 keys per sequence
constexpr int kG    = kHq / kHkv;         // 4
constexpr int kChunk = 32;                // keys per inner iteration
constexpr int kRowsTotal = kB * kHkv * kG * kQ;   // 16384 softmax rows
constexpr int KROW = 132;                 // padded f32 per K row in LDS (528B, 16B aligned)
constexpr int VROW = 40;                  // padded bf16 per V^T row (80B)
constexpr int PROW = 40;                  // padded bf16 per P row (80B)
constexpr float kScaleLog2e = 0.08838834764831845f * 1.4426950408889634f;
}

// ---------------- async global->LDS staging (CDNA5 ASYNCcnt path) -------------
DEVFN void async_copy_b128(const float* g, float* lds) {
#if __has_builtin(__builtin_amdgcn_global_load_async_to_lds_b128)
  __builtin_amdgcn_global_load_async_to_lds_b128(
      (__attribute__((address_space(1))) v4i*)g,
      (__attribute__((address_space(3))) v4i*)lds, 0, 0);
#else
  *(f32x4*)lds = *(const f32x4*)g;   // synchronous fallback
#endif
}
DEVFN void wait_async_le8() {
#if __has_builtin(__builtin_amdgcn_global_load_async_to_lds_b128)
  asm volatile("s_wait_asynccnt 0x8" ::: "memory");
#endif
}
DEVFN void wait_async_le0() {
#if __has_builtin(__builtin_amdgcn_global_load_async_to_lds_b128)
  asm volatile("s_wait_asynccnt 0x0" ::: "memory");
#endif
}

DEVFN v8f wmma_bf16(v16bf a, v16bf b, v8f c) {
  return __builtin_amdgcn_wmma_f32_16x16x32_bf16(false, a, false, b, (short)0, c,
                                                 false, false);
}

// Build a 16-element bf16 B/K fragment run from 16 contiguous f32 in LDS.
DEVFN v16bf frag_from_f32(const float* p) {
  f32x4 x0 = *(const f32x4*)(p);
  f32x4 x1 = *(const f32x4*)(p + 4);
  f32x4 x2 = *(const f32x4*)(p + 8);
  f32x4 x3 = *(const f32x4*)(p + 12);
  v16bf r;
#pragma unroll
  for (int j = 0; j < 4; ++j) {
    r[j]      = (__bf16)x0[j];
    r[j + 4]  = (__bf16)x1[j];
    r[j + 8]  = (__bf16)x2[j];
    r[j + 12] = (__bf16)x3[j];
  }
  return r;
}

// ---------------------------- main attention kernel ---------------------------
// Grid: (B*Hkv, splits).  Block: 128 threads = 4 waves; wave w owns GQA head
// g = w (16 query rows).  Flash-attention over 32-key chunks:
//   S (16x32) = Q_bf16 x K_bf16^T   (8 wmma)  -> online softmax (exp2 domain)
//   O (16x128) += P_bf16 x V_bf16   (8 wmma)
//   l (row sums) += P_bf16 x ones   (1 wmma, replaces shuffle-reduce)
template <bool DIRECT>
__global__ __launch_bounds__(128) void fa_kernel(
    const float* __restrict__ query, const float* __restrict__ key_cache,
    const float* __restrict__ value_cache, const int* __restrict__ page_table,
    float* __restrict__ opart,   // DIRECT: final out [B*Q][Hq*D]; else [splits][16384][128]
    float* __restrict__ mpart,   // [splits][16384]
    float* __restrict__ lpart,   // [splits][16384]
    int keys_per_split) {
  __shared__ __attribute__((aligned(16))) float  Ks[2][kChunk][KROW];  // f32, async-staged
  __shared__ __attribute__((aligned(16))) __bf16 VsT[kD][VROW];        // V^T bf16
  __shared__ __attribute__((aligned(16))) __bf16 Ps[4][kQ][PROW];      // per-wave P tile

  const int bh    = blockIdx.x;
  const int b     = bh / kHkv;
  const int hkv   = bh % kHkv;
  const int split = blockIdx.y;
  const int tid   = threadIdx.x;
  const int w     = tid >> 5;    // wave = GQA group member g
  const int lane  = tid & 31;
  const int n     = lane & 15;   // column / row-within-tile index
  const int hs    = lane >> 4;   // lane-half select

  const int key0   = split * keys_per_split;
  const int nchunk = keys_per_split / kChunk;

  // ---- Q fragments (A-layout: lane holds row n, one contiguous 16-elem K run), scale folded
  v16bf qa[4];
  {
    const float* qb =
        query + (((size_t)b * kQ + n) * kHq + (hkv * kG + w)) * kD;
#pragma unroll
    for (int s = 0; s < 4; ++s) {
      const int d0 = s * 32 + hs * 16;
      f32x4 x0 = *(const f32x4*)(qb + d0);
      f32x4 x1 = *(const f32x4*)(qb + d0 + 4);
      f32x4 x2 = *(const f32x4*)(qb + d0 + 8);
      f32x4 x3 = *(const f32x4*)(qb + d0 + 12);
#pragma unroll
      for (int j = 0; j < 4; ++j) {
        qa[s][j]      = (__bf16)(x0[j] * kScaleLog2e);
        qa[s][j + 4]  = (__bf16)(x1[j] * kScaleLog2e);
        qa[s][j + 8]  = (__bf16)(x2[j] * kScaleLog2e);
        qa[s][j + 12] = (__bf16)(x3[j] * kScaleLog2e);
      }
    }
  }

  v16bf ones;
#pragma unroll
  for (int j = 0; j < 16; ++j) ones[j] = (__bf16)1.0f;

  const v8f vzero = {0.f, 0.f, 0.f, 0.f, 0.f, 0.f, 0.f, 0.f};
  v8f o[8];
  v8f lacc = vzero;            // row sums via ones-column WMMA
  float mrow[8];
#pragma unroll
  for (int t = 0; t < 8; ++t) o[t] = vzero;
#pragma unroll
  for (int v = 0; v < 8; ++v) mrow[v] = -1e30f;

  // Async K staging: each wave-instruction copies one full 512B K row (coalesced).
  auto stage_k_async = [&](int chunk, int buf) {
#pragma unroll
    for (int i = 0; i < 8; ++i) {
      const int kk = w + 4 * i;
      const int ka = key0 + chunk * kChunk + kk;
      const int pg = page_table[b * kPPS + (ka >> 4)];
      const float* g = key_cache +
          (((size_t)pg * kPage + (ka & 15)) * kHkv + hkv) * kD + lane * 4;
      async_copy_b128(g, &Ks[buf][kk][lane * 4]);
    }
  };
  // V staging: coalesced f32 loads, cvt to bf16, transposed scatter into LDS.
  auto stage_v = [&](int chunk) {
#pragma unroll
    for (int i = 0; i < 8; ++i) {
      const int kk = w + 4 * i;
      const int ka = key0 + chunk * kChunk + kk;
      const int pg = page_table[b * kPPS + (ka >> 4)];
      f32x4 x = *(const f32x4*)(value_cache +
          (((size_t)pg * kPage + (ka & 15)) * kHkv + hkv) * kD + lane * 4);
#pragma unroll
      for (int j = 0; j < 4; ++j) VsT[lane * 4 + j][kk] = (__bf16)x[j];
    }
  };

  stage_k_async(0, 0);

  for (int c = 0; c < nchunk; ++c) {
    const int buf = c & 1;
    stage_v(c);
    const bool hasNext = (c + 1 < nchunk);
    if (hasNext) stage_k_async(c + 1, buf ^ 1);
    if (hasNext) wait_async_le8(); else wait_async_le0();  // chunk c's asyncs done
    __syncthreads();

    const int kbase = key0 + c * kChunk;

    // ---- S = Q K^T (two 16x16 tiles) ----
    v8f sc[2];
#pragma unroll
    for (int ct = 0; ct < 2; ++ct) {
      v8f acc = vzero;
      const float* krow = &Ks[buf][ct * 16 + n][0];   // B-layout: lane -> key row
#pragma unroll
      for (int s = 0; s < 4; ++s) {
        v16bf kf = frag_from_f32(krow + s * 32 + hs * 16);
        acc = wmma_bf16(qa[s], kf, acc);
      }
      sc[ct] = acc;
    }

    // ---- causal mask (only the final chunk of the context) ----
    if (kbase + kChunk > kKtot - kQ) {
#pragma unroll
      for (int ct = 0; ct < 2; ++ct) {
        const int ka = kbase + ct * 16 + n;
#pragma unroll
        for (int v = 0; v < 8; ++v)
          if (ka > kKtot - kQ + (v + 8 * hs)) sc[ct][v] = -3.0e38f;
      }
    }

    // ---- online softmax over this 32-key chunk (exp2 domain) ----
    float al[8];
#pragma unroll
    for (int v = 0; v < 8; ++v) {
      float mx = fmaxf(sc[0][v], sc[1][v]);
#pragma unroll
      for (int off = 8; off >= 1; off >>= 1)
        mx = fmaxf(mx, __shfl_xor(mx, off, 16));
      const float mn = fmaxf(mrow[v], mx);
      al[v] = exp2f(mrow[v] - mn);
      mrow[v] = mn;
      sc[0][v] = exp2f(sc[0][v] - mn);
      sc[1][v] = exp2f(sc[1][v] - mn);
    }
#pragma unroll
    for (int t = 0; t < 8; ++t)
#pragma unroll
      for (int v = 0; v < 8; ++v) o[t][v] *= al[v];
#pragma unroll
    for (int v = 0; v < 8; ++v) lacc[v] *= al[v];

    // ---- P (C-layout f32) -> LDS bf16, reload in A-layout ----
#pragma unroll
    for (int v = 0; v < 8; ++v) {
      const int r = v + 8 * hs;
      Ps[w][r][n]      = (__bf16)sc[0][v];
      Ps[w][r][n + 16] = (__bf16)sc[1][v];
    }
    v16bf pa;
    {
      const v8bf lo = *(const v8bf*)&Ps[w][n][hs * 8];
      const v8bf hi = *(const v8bf*)&Ps[w][n][hs * 8 + 16];
#pragma unroll
      for (int j = 0; j < 8; ++j) { pa[j] = lo[j]; pa[j + 8] = hi[j]; }
    }

    // ---- O += P V  (8 n-tiles);  l += P * ones (row-sum tile) ----
    lacc = wmma_bf16(pa, ones, lacc);
#pragma unroll
    for (int t = 0; t < 8; ++t) {
      v16bf vf;
      const v8bf a0 = *(const v8bf*)&VsT[t * 16 + n][hs * 8];
      const v8bf a1 = *(const v8bf*)&VsT[t * 16 + n][hs * 8 + 16];
#pragma unroll
      for (int j = 0; j < 8; ++j) { vf[j] = a0[j]; vf[j + 8] = a1[j]; }
      o[t] = wmma_bf16(pa, vf, o[t]);
    }
    __syncthreads();  // VsT may be restaged next iteration
  }

  // ---- epilogue ----
  const int h = hkv * kG + w;
  if constexpr (DIRECT) {
#pragma unroll
    for (int v = 0; v < 8; ++v) {
      const int q = v + 8 * hs;
      const float inv = 1.0f / lacc[v];
      float* ob = opart + (((size_t)b * kQ + q) * kHq + h) * kD + n;
#pragma unroll
      for (int t = 0; t < 8; ++t) ob[t * 16] = o[t][v] * inv;
    }
  } else {
    const int rbase = (bh * kG + w) * kQ;
#pragma unroll
    for (int v = 0; v < 8; ++v) {
      const int row = rbase + v + 8 * hs;
      float* ob = opart + ((size_t)split * kRowsTotal + row) * kD + n;
#pragma unroll
      for (int t = 0; t < 8; ++t) ob[t * 16] = o[t][v];
      if (n == 0) {
        mpart[(size_t)split * kRowsTotal + row] = mrow[v];
        lpart[(size_t)split * kRowsTotal + row] = lacc[v];
      }
    }
  }
}

// -------------------- split-K combine (flash-decoding merge) ------------------
__global__ __launch_bounds__(256) void combine_kernel(
    const float* __restrict__ opart, const float* __restrict__ mpart,
    const float* __restrict__ lpart, float* __restrict__ out, int splits) {
  const int idx = blockIdx.x * blockDim.x + threadIdx.x;
  if (idx >= kB * kQ * kHq * kD) return;
  const int d  = idx & (kD - 1);
  const int h  = (idx >> 7) & (kHq - 1);
  const int tq = idx >> 12;  // b*16+q
  const int b  = tq >> 4;
  const int q  = tq & 15;
  const int row = ((b * kHkv + h / kG) * kG + (h % kG)) * kQ + q;
  float mg = -1e30f;
  for (int s = 0; s < splits; ++s) mg = fmaxf(mg, mpart[s * kRowsTotal + row]);
  float den = 0.f, num = 0.f;
  for (int s = 0; s < splits; ++s) {
    const float wgt = exp2f(mpart[s * kRowsTotal + row] - mg);
    den += lpart[s * kRowsTotal + row] * wgt;
    num += opart[((size_t)s * kRowsTotal + row) * kD + d] * wgt;
  }
  out[idx] = num / den;
}

extern "C" void kernel_launch(void* const* d_in, const int* in_sizes, int n_in,
                              void* d_out, int out_size, void* d_ws,
                              size_t ws_size, hipStream_t stream) {
  (void)in_sizes; (void)n_in; (void)out_size;
  const float* query = (const float*)d_in[0];
  const float* keyc  = (const float*)d_in[1];
  const float* valc  = (const float*)d_in[2];
  const int*   pt    = (const int*)d_in[3];
  float* out = (float*)d_out;

  auto need = [](int s) -> size_t {
    return (size_t)s * kRowsTotal * kD * sizeof(float) +
           2u * (size_t)s * kRowsTotal * sizeof(float);
  };
  int splits = 0;
  if (ws_size >= need(4)) splits = 4;
  else if (ws_size >= need(2)) splits = 2;
  else if (ws_size >= need(1)) splits = 1;

  dim3 blk(128);
  if (splits > 0) {
    float* opart = (float*)d_ws;
    float* mpart = opart + (size_t)splits * kRowsTotal * kD;
    float* lpart = mpart + (size_t)splits * kRowsTotal;
    dim3 grid(kB * kHkv, splits);
    fa_kernel<false><<<grid, blk, 0, stream>>>(query, keyc, valc, pt, opart,
                                               mpart, lpart, kKtot / splits);
    const int total = kB * kQ * kHq * kD;
    combine_kernel<<<(total + 255) / 256, 256, 0, stream>>>(opart, mpart, lpart,
                                                            out, splits);
  } else {
    dim3 grid(kB * kHkv, 1);
    fa_kernel<true><<<grid, blk, 0, stream>>>(query, keyc, valc, pt, out,
                                              nullptr, nullptr, kKtot);
  }
}
